// HGAT_11562051961295
// MI455X (gfx1250) — matrix operand order repaired and checked by reading, not compiled, not run on hardware
//
#include <hip/hip_runtime.h>
#include <hip/hip_bf16.h>

// ---------------- types / helpers ----------------
typedef __bf16 bf16;
typedef bf16         v16bf __attribute__((ext_vector_type(16)));
typedef float        v8f   __attribute__((ext_vector_type(8)));
typedef float        v4f   __attribute__((ext_vector_type(4)));
typedef unsigned int v4u   __attribute__((ext_vector_type(4)));
typedef unsigned int u32x4 __attribute__((ext_vector_type(4)));
typedef int          i32x8 __attribute__((ext_vector_type(8)));
typedef int          i32x4 __attribute__((ext_vector_type(4)));

__device__ __forceinline__ unsigned short f32_to_bf16_bits(float f) {
    unsigned u = __builtin_bit_cast(unsigned, f);
    u += 0x7FFFu + ((u >> 16) & 1u);          // round-to-nearest-even
    return (unsigned short)(u >> 16);
}

#define NTYPE 3
#define F_DIM 512
#define HID   512
#define C2    18
#define ATT_N 50
#define LEAKY 0.2f

// Tensor Data Mover availability (this toolchain: 6-arg builtin,
//   (u32x4 g0, i32x8 g1, i32x4 g2, i32x4 g3, i32x8, i32 cpol))
#if defined(__HIP_DEVICE_COMPILE__) && __has_builtin(__builtin_amdgcn_tensor_load_to_lds) && \
    __has_builtin(__builtin_amdgcn_s_wait_tensorcnt)
#define USE_TDM 1
#else
#define USE_TDM 0
#endif

// ---------------- GEMM1: support1[t] = xs[t] @ W1[t]  (bf16 WMMA, f32 acc) ----------------
#define TILE_M 64
#define TILE_N 32
#define TILE_K 32
#define A_STRIDE 36   // floats per LDS row: 32 data + 4 pad dwords (16B-aligned, bank-spread)
#define B_STRIDE 40   // bf16 per LDS row (80B = 16*5)

__global__ void hgat_gemm1_wmma(const float* __restrict__ A,   // [T, M, K]
                                const float* __restrict__ B,   // [T, K, Nn]
                                float* __restrict__ C,         // [T, M, Nn]
                                int M, int K, int Nn) {
    __shared__ __align__(16) float          Af32[TILE_M][A_STRIDE]; // A tile, f32 row-major
    __shared__ __align__(16) unsigned short BsT[TILE_N][B_STRIDE];  // B tile, bf16 transposed [n][k]

    const int t  = blockIdx.z;
    const int m0 = blockIdx.y * TILE_M;
    const int n0 = blockIdx.x * TILE_N;
    const float* Ag = A + (size_t)t * M * K;
    const float* Bg = B + (size_t)t * K * Nn;
    float*       Cg = C + (size_t)t * M * Nn;

    const int tid   = threadIdx.x;        // 128 threads = 4 waves
    const int wave  = tid >> 5;
    const int lane  = tid & 31;
    const int lhalf = lane >> 4;          // 0: lanes 0-15, 1: lanes 16-31
    const int lr    = lane & 15;

    v8f acc0 = {};
    v8f acc1 = {};

    for (int k0 = 0; k0 < K; k0 += TILE_K) {
        // ---- stage A tile (64x32 f32) into LDS ----
#if USE_TDM
        if (wave == 0) {
            // D# group 0: count=1, lds_addr, 57-bit global addr, type=2
            unsigned ldsAddr = (unsigned)(size_t)&Af32[0][0];
            unsigned long long ga = (unsigned long long)(size_t)(Ag + (size_t)m0 * K + k0);
            u32x4 g0;
            g0[0] = 1u;                                   // count=1, user flags 0
            g0[1] = ldsAddr;                              // [63:32]  lds_addr
            g0[2] = (unsigned)ga;                         // [95:64]  global_addr lo
            g0[3] = (unsigned)((ga >> 32) & 0x01FFFFFFu)  // [120:96] global_addr hi
                    | (2u << 30);                         // [127:126] type=2 (image)
            // D# group 1
            unsigned rowsRem = (unsigned)(M - m0);        // tensor_dim1: OOB rows -> zero fill
            i32x8 g1;
            g1[0] = (int)((2u << 16)        // data_size = 4 bytes
                        | (1u << 20)        // pad_enable
                        | (4u << 22)        // pad_interval: 32 dwords
                        | (3u << 25));      // pad_amount: 4 dwords  -> 36-dword LDS rows
            g1[1] = (int)(((unsigned)K & 0xFFFFu) << 16);            // tensor_dim0[15:0]
            g1[2] = (int)((((unsigned)K >> 16) & 0xFFFFu)            // tensor_dim0[31:16]
                        | ((rowsRem & 0xFFFFu) << 16));              // tensor_dim1[15:0]
            g1[3] = (int)(((rowsRem >> 16) & 0xFFFFu)                // tensor_dim1[31:16]
                        | ((unsigned)TILE_K << 16));                 // tile_dim0 = 32
            g1[4] = (int)TILE_M;                                     // tile_dim1 = 64
            g1[5] = (int)K;                                          // tensor_dim0_stride lo
            g1[6] = 0;                                               // stride hi / dim1_stride lo
            g1[7] = 0;
            i32x4 gz4 = {0, 0, 0, 0};
            i32x8 gz8 = {0, 0, 0, 0, 0, 0, 0, 0};
            __builtin_amdgcn_tensor_load_to_lds(g0, g1, gz4, gz4, gz8, 0);
        }
#else
        #pragma unroll
        for (int i = 0; i < (TILE_M * TILE_K) / (128 * 4); ++i) {   // 4 iters of float4
            int chunk = tid + i * 128;          // 512 float4 chunks
            int r  = chunk >> 3;                // row in tile
            int c4 = (chunk & 7) * 4;           // starting col
            int gr = m0 + r;
            if (gr >= M) gr = M - 1;            // clamp (stores are guarded)
            v4f v = *(const v4f*)&Ag[(size_t)gr * K + k0 + c4];
            *(v4f*)&Af32[r][c4] = v;
        }
#endif
        // ---- stage B tile (32x32 f32 -> bf16, transposed [n][k]) ----
        #pragma unroll
        for (int i = 0; i < (TILE_K * TILE_N) / (128 * 4); ++i) {   // 2 iters of float4
            int chunk = tid + i * 128;          // 256 float4 chunks
            int r  = chunk >> 3;                // k row
            int c4 = (chunk & 7) * 4;           // starting n col
            v4f v = *(const v4f*)&Bg[(size_t)(k0 + r) * Nn + n0 + c4];
            BsT[c4 + 0][r] = f32_to_bf16_bits(v[0]);
            BsT[c4 + 1][r] = f32_to_bf16_bits(v[1]);
            BsT[c4 + 2][r] = f32_to_bf16_bits(v[2]);
            BsT[c4 + 3][r] = f32_to_bf16_bits(v[3]);
        }
        // prefetch next K-tile of B (global_prefetch_b8)
        if (k0 + TILE_K < K)
            __builtin_prefetch(Bg + (size_t)(k0 + TILE_K + wave) * Nn + n0 + lane, 0, 1);
#if USE_TDM
        __builtin_amdgcn_s_wait_tensorcnt(0);
#endif
        __syncthreads();

        // ---- A fragment 16x32 bf16: lanes 0-15 K{0..15}; lanes 16-31 K{16..31} ----
        const int arow  = (wave << 4) + lr;
        const int kbase = lhalf * 16;
        v4f a0 = *(const v4f*)&Af32[arow][kbase];
        v4f a1 = *(const v4f*)&Af32[arow][kbase + 4];
        v4f a2 = *(const v4f*)&Af32[arow][kbase + 8];
        v4f a3 = *(const v4f*)&Af32[arow][kbase + 12];
        union { v16bf v; unsigned short s[16]; } af;
        #pragma unroll
        for (int j = 0; j < 4; ++j) {
            af.s[j]      = f32_to_bf16_bits(a0[j]);
            af.s[4 + j]  = f32_to_bf16_bits(a1[j]);
            af.s[8 + j]  = f32_to_bf16_bits(a2[j]);
            af.s[12 + j] = f32_to_bf16_bits(a3[j]);
        }
        // ---- B fragments 32x16 bf16: contiguous 32B per lane from transposed stage ----
        union { v16bf v; v4u q[2]; } bf0, bf1;
        bf0.q[0] = *(const v4u*)&BsT[lr][lhalf * 16];
        bf0.q[1] = *(const v4u*)&BsT[lr][lhalf * 16 + 8];
        bf1.q[0] = *(const v4u*)&BsT[16 + lr][lhalf * 16];
        bf1.q[1] = *(const v4u*)&BsT[16 + lr][lhalf * 16 + 8];

        acc0 = __builtin_amdgcn_wmma_f32_16x16x32_bf16(false, af.v, false, bf0.v,
                                                       (short)0, acc0, false, false);
        acc1 = __builtin_amdgcn_wmma_f32_16x16x32_bf16(false, af.v, false, bf1.v,
                                                       (short)0, acc1, false, false);
        __syncthreads();
    }

    // C layout: VGPR r -> M=r (lanes 0-15), M=r+8 (lanes 16-31); N = lane%16
    #pragma unroll
    for (int r = 0; r < 8; ++r) {
        int grow = m0 + (wave << 4) + r + (lhalf ? 8 : 0);
        if (grow < M) {
            Cg[(size_t)grow * Nn + n0 + lr]      = acc0[r];
            Cg[(size_t)grow * Nn + n0 + 16 + lr] = acc1[r];
        }
    }
}

// ---------------- init: dst[n*3*D + i] = bias[i % D] ----------------
__global__ void hgat_init_bias(float* __restrict__ dst, const float* __restrict__ bias,
                               int D, long long total) {
    long long g = (long long)blockIdx.x * blockDim.x + threadIdx.x;
    if (g >= total) return;
    dst[g] = bias[(int)(g % D)];
}

// ---------------- SpMM: dst[r*stride + f] += v * src[c*D + f] (edge-parallel atomics) ----------------
__global__ void hgat_spmm(const int* __restrict__ rows, const int* __restrict__ cols,
                          const float* __restrict__ vals, const float* __restrict__ src,
                          float* __restrict__ dst, long long total, int D, int dstStride) {
    long long g = (long long)blockIdx.x * blockDim.x + threadIdx.x;
    if (g >= total) return;
    int e = (int)(g / D);
    int f = (int)(g % D);
    int r = rows[e];
    int c = cols[e];
    float v = vals[e];
    __hip_atomic_fetch_add(dst + (size_t)r * dstStride + f, v * src[(size_t)c * D + f],
                           __ATOMIC_RELAXED, __HIP_MEMORY_SCOPE_AGENT);
}

// ---------------- attention projection: wlo = W@a[:ATT], whi = W@a[ATT:], c = b.(a_lo+a_hi) ----------------
__global__ void hgat_proj(const float* __restrict__ W,  // [D, ATT]
                          const float* __restrict__ b,  // [ATT]
                          const float* __restrict__ a,  // [2*ATT]
                          float* __restrict__ wlo, float* __restrict__ whi,
                          float* __restrict__ cout, int D) {
    int k = blockIdx.x * blockDim.x + threadIdx.x;
    if (k < D) {
        float lo = 0.f, hi = 0.f;
        for (int j = 0; j < ATT_N; ++j) {
            float w = W[k * ATT_N + j];
            lo += w * a[j];
            hi += w * a[ATT_N + j];
        }
        wlo[k] = lo;
        whi[k] = hi;
    }
    if (k == 0) {
        float c = 0.f;
        for (int j = 0; j < ATT_N; ++j) c += b[j] * (a[j] + a[ATT_N + j]);
        *cout = c;
    }
}

// ---------------- layer-1 attention + relu (wave per node, D=512) ----------------
__global__ void hgat_attn1(const float* __restrict__ hs,  // [N,3,512]
                           const float* __restrict__ wlo, const float* __restrict__ whi,
                           const float* __restrict__ cptr, int idx,
                           float* __restrict__ out, int nNodes) {
    int wid  = (int)((blockIdx.x * (long long)blockDim.x + threadIdx.x) >> 5);
    int lane = threadIdx.x & 31;
    if (wid >= nNodes) return;
    const float* base = hs + (size_t)wid * 3 * HID;
    float cadd = *cptr;

    float dhi[NTYPE];
    float dlo = 0.f;
    for (int t = 0; t < NTYPE; ++t) {
        float ph = 0.f, pl = 0.f;
        #pragma unroll
        for (int i = 0; i < HID / 32; ++i) {
            int k = lane + 32 * i;
            float v = base[t * HID + k];
            ph += v * whi[k];
            if (t == idx) pl += v * wlo[k];
        }
        for (int off = 16; off; off >>= 1) {
            ph += __shfl_xor(ph, off, 32);
            if (t == idx) pl += __shfl_xor(pl, off, 32);
        }
        dhi[t] = ph;
        if (t == idx) dlo = pl;
    }
    float e[NTYPE], m = -1e30f;
    for (int t = 0; t < NTYPE; ++t) {
        float x = dlo + dhi[t] + cadd;
        e[t] = (x > 0.f) ? x : LEAKY * x;
        m = fmaxf(m, e[t]);
    }
    float s = 0.f, w[NTYPE];
    for (int t = 0; t < NTYPE; ++t) { w[t] = __expf(e[t] - m); s += w[t]; }
    float inv = 1.f / s;
    #pragma unroll
    for (int i = 0; i < HID / 32; ++i) {
        int k = lane + 32 * i;
        float acc = 0.f;
        for (int t = 0; t < NTYPE; ++t) acc += w[t] * inv * base[t * HID + k];
        out[(size_t)wid * HID + k] = fmaxf(acc, 0.f);   // relu
    }
}

// ---------------- GEMM2: support2[t] = x1[t] @ W2 (512 -> 18), wave per node, W2 in LDS ----------------
__global__ void hgat_gemm2(const float* __restrict__ x1,  // [3, N, 512]
                           const float* __restrict__ W2,  // [512, 18]
                           float* __restrict__ s2,        // [3, N, 18]
                           int nNodes) {
    __shared__ float Ws[HID * C2];
    for (int i = threadIdx.x; i < HID * C2; i += blockDim.x) Ws[i] = W2[i];
    __syncthreads();

    const int t    = blockIdx.z;
    const int wpb  = blockDim.x >> 5;
    const int node = blockIdx.x * wpb + (threadIdx.x >> 5);
    const int lane = threadIdx.x & 31;
    if (node >= nNodes) return;

    const float* xr = x1 + ((size_t)t * nNodes + node) * HID;
    float acc[C2];
    #pragma unroll
    for (int c = 0; c < C2; ++c) acc[c] = 0.f;
    #pragma unroll
    for (int i = 0; i < HID / 32; ++i) {
        int k = lane + 32 * i;
        float xv = xr[k];
        #pragma unroll
        for (int c = 0; c < C2; ++c) acc[c] += xv * Ws[k * C2 + c];
    }
    #pragma unroll
    for (int c = 0; c < C2; ++c)
        for (int off = 16; off; off >>= 1) acc[c] += __shfl_xor(acc[c], off, 32);
    if (lane == 0) {
        float* o = s2 + ((size_t)t * nNodes + node) * C2;
        #pragma unroll
        for (int c = 0; c < C2; ++c) o[c] = acc[c];
    }
}

// ---------------- layer-2 attention + log_softmax (thread per node, D=18) ----------------
__global__ void hgat_attn2(const float* __restrict__ hs2,  // [N,3,18]
                           const float* __restrict__ wlo, const float* __restrict__ whi,
                           const float* __restrict__ cptr, int idx,
                           float* __restrict__ out, int nNodes) {
    int n = blockIdx.x * blockDim.x + threadIdx.x;
    if (n >= nNodes) return;
    const float* base = hs2 + (size_t)n * 3 * C2;
    float cadd = *cptr;

    float dlo = 0.f;
    #pragma unroll
    for (int k = 0; k < C2; ++k) dlo += base[idx * C2 + k] * wlo[k];

    float e[NTYPE], m = -1e30f;
    for (int t = 0; t < NTYPE; ++t) {
        float dh = 0.f;
        #pragma unroll
        for (int k = 0; k < C2; ++k) dh += base[t * C2 + k] * whi[k];
        float x = dlo + dh + cadd;
        e[t] = (x > 0.f) ? x : LEAKY * x;
        m = fmaxf(m, e[t]);
    }
    float s = 0.f, w[NTYPE];
    for (int t = 0; t < NTYPE; ++t) { w[t] = __expf(e[t] - m); s += w[t]; }
    float inv = 1.f / s;

    float h[C2], hm = -1e30f;
    #pragma unroll
    for (int k = 0; k < C2; ++k) {
        float acc = 0.f;
        for (int t = 0; t < NTYPE; ++t) acc += w[t] * inv * base[t * C2 + k];
        h[k] = acc;
        hm = fmaxf(hm, acc);
    }
    float se = 0.f;
    #pragma unroll
    for (int k = 0; k < C2; ++k) se += __expf(h[k] - hm);
    float lse = hm + __logf(se);
    #pragma unroll
    for (int k = 0; k < C2; ++k) out[(size_t)n * C2 + k] = h[k] - lse;
}

// ---------------- orchestration ----------------
extern "C" void kernel_launch(void* const* d_in, const int* in_sizes, int n_in,
                              void* d_out, int out_size, void* d_ws, size_t ws_size,
                              hipStream_t stream) {
    const float* xs    = (const float*)d_in[0];   // [3, N, 512]
    const int*   rows  = (const int*)  d_in[1];   // [3, 3, E]
    const int*   cols  = (const int*)  d_in[2];   // [3, 3, E]
    const float* vals  = (const float*)d_in[3];   // [3, 3, E]
    const float* W1    = (const float*)d_in[4];   // [3, 512, 512]
    const float* bias1 = (const float*)d_in[5];   // [512]
    const float* W2    = (const float*)d_in[6];   // [512, 18]
    const float* b2    = (const float*)d_in[7];   // [18]
    const float* at1W  = (const float*)d_in[8];   // [3, 512, 50]
    const float* at1b  = (const float*)d_in[9];   // [3, 50]
    const float* at1a  = (const float*)d_in[10];  // [3, 100]
    const float* at2W  = (const float*)d_in[11];  // [3, 18, 50]
    const float* at2b  = (const float*)d_in[12];  // [3, 50]
    const float* at2a  = (const float*)d_in[13];  // [3, 100]
    float* outp = (float*)d_out;                  // [3, N, 18]

    const int N = in_sizes[0] / (NTYPE * F_DIM);  // 50000
    const int E = in_sizes[1] / (NTYPE * NTYPE);  // 1600000

    // workspace carve-up (floats)
    float* ws = (float*)d_ws;
    size_t off = 0;
    float* support1 = ws + off; off += (size_t)NTYPE * N * HID;   // [3,N,512]
    float* x1       = ws + off; off += (size_t)NTYPE * N * HID;   // [3,N,512]
    float* support2 = ws + off; off += (size_t)NTYPE * N * C2;    // [3,N,18]
    float* hs       = ws + off; off += (size_t)N * NTYPE * HID;   // [N,3,512] (reused per t1)
    float* hs2      = ws + off; off += (size_t)N * NTYPE * C2;    // [N,3,18]  (reused per t1)
    float* wlo      = ws + off; off += HID;
    float* whi      = ws + off; off += HID;
    float* cbuf     = ws + off; off += 1;

    // ---- layer 1 dense: support1 = xs @ W1 (bf16 WMMA + TDM staging) ----
    {
        dim3 grid(HID / TILE_N, (N + TILE_M - 1) / TILE_M, NTYPE);
        hgat_gemm1_wmma<<<grid, 128, 0, stream>>>(xs, W1, support1, N, F_DIM, HID);
    }

    // ---- layer 1 per target type ----
    for (int t1 = 0; t1 < NTYPE; ++t1) {
        long long tot1 = (long long)N * NTYPE * HID;
        hgat_init_bias<<<(unsigned)((tot1 + 255) / 256), 256, 0, stream>>>(hs, bias1, HID, tot1);
        for (int t2 = 0; t2 < NTYPE; ++t2) {
            long long tote = (long long)E * HID;
            const size_t eo = ((size_t)t1 * NTYPE + t2) * E;
            hgat_spmm<<<(unsigned)((tote + 255) / 256), 256, 0, stream>>>(
                rows + eo, cols + eo, vals + eo,
                support1 + (size_t)t2 * N * HID,
                hs + (size_t)t2 * HID, tote, HID, NTYPE * HID);
        }
        hgat_proj<<<(HID + 255) / 256, 256, 0, stream>>>(
            at1W + (size_t)t1 * HID * ATT_N, at1b + (size_t)t1 * ATT_N,
            at1a + (size_t)t1 * 2 * ATT_N, wlo, whi, cbuf, HID);
        int wavesPerBlock = 256 / 32;
        hgat_attn1<<<(N + wavesPerBlock - 1) / wavesPerBlock, 256, 0, stream>>>(
            hs, wlo, whi, cbuf, t1, x1 + (size_t)t1 * N * HID, N);
    }

    // ---- layer 2 dense: support2 = x1 @ W2 ----
    {
        int wavesPerBlock = 256 / 32;
        dim3 grid((N + wavesPerBlock - 1) / wavesPerBlock, 1, NTYPE);
        hgat_gemm2<<<grid, 256, 0, stream>>>(x1, W2, support2, N);
    }

    // ---- layer 2 per target type ----
    for (int t1 = 0; t1 < NTYPE; ++t1) {
        long long tot2 = (long long)N * NTYPE * C2;
        hgat_init_bias<<<(unsigned)((tot2 + 255) / 256), 256, 0, stream>>>(hs2, b2, C2, tot2);
        for (int t2 = 0; t2 < NTYPE; ++t2) {
            long long tote = (long long)E * C2;
            const size_t eo = ((size_t)t1 * NTYPE + t2) * E;
            hgat_spmm<<<(unsigned)((tote + 255) / 256), 256, 0, stream>>>(
                rows + eo, cols + eo, vals + eo,
                support2 + (size_t)t2 * N * C2,
                hs2 + (size_t)t2 * C2, tote, C2, NTYPE * C2);
        }
        hgat_proj<<<1, 256, 0, stream>>>(
            at2W + (size_t)t1 * C2 * ATT_N, at2b + (size_t)t1 * ATT_N,
            at2a + (size_t)t1 * 2 * ATT_N, wlo, whi, cbuf, C2);
        hgat_attn2<<<(N + 255) / 256, 256, 0, stream>>>(
            hs2, wlo, whi, cbuf, t1, outp + (size_t)t1 * N * C2, N);
    }
    (void)n_in; (void)out_size; (void)ws_size;
}
